// ODERNNEncoder_76493367542049
// MI455X (gfx1250) — matrix operand order repaired and checked by reading, not compiled, not run on hardware
//
#include <hip/hip_runtime.h>

typedef __attribute__((ext_vector_type(8)))  float  v8f;
typedef __attribute__((ext_vector_type(16))) __bf16 v16bf;

#define T_STEPS 128
#define OBS_D   64
#define NODE    20
#define B_TOTAL 4096

// workspace byte offsets (all 1KB-aligned)
#define OFF_W1T    0u        // 64 frags  (Mtiles=16, Ktiles=4)
#define OFF_W2T    65536u    // 64 frags  (8,8)
#define OFF_WIH    131072u   // 96 frags  (24,4)
#define OFF_WHH    229376u   // 96 frags  (24,4)
#define OFF_WEMBT  327680u   // 16 frags  (8,2)
#define OFF_WPROJT 344064u   // 16 frags  (4,4)
#define OFF_B1F    360448u   // 16 bias tiles (f32)
#define OFF_B2F    376832u   // 8
#define OFF_BIHF   385024u   // 24
#define OFF_BHHF   409600u   // 24
#define OFF_BEMBF  434176u   // 8
#define OFF_BPROJF 442368u   // 4

// LDS (bytes): w1T 0..64K, w2T 64..128K, RK4 acc (per-wave) 128..144K,
// partial-kbuf exchange (per-wave) 144..160K  => 163840 total (2 WG/WGP possible)
#define LDS_W2   65536
#define LDS_ACC  131072
#define LDS_XCH  147456
#define LDS_SIZE 163840

__device__ __forceinline__ float swapx16(float x) {
  // ds_swizzle group-of-32: xor=0x10, or=0, and=0x1f -> swap lanes 0-15 <-> 16-31
  return __int_as_float(__builtin_amdgcn_ds_swizzle(__float_as_int(x), 0x401F));
}

__device__ __forceinline__ v8f wmma_bf16(v16bf a, v16bf b, v8f c) {
  return __builtin_amdgcn_wmma_f32_16x16x32_bf16(false, a, false, b, (short)0, c, false, false);
}

// Build a bf16 B-fragment (K=32 x N=16) from two f32 D-tiles (K 0..15 / 16..31).
// B layout: lane half0 holds K = j, half1 holds K = j+16 (j = element 0..15).
// D layout: lane half0 holds M = e (0..7), half1 holds M = 8+e.
__device__ __forceinline__ v16bf make_bfrag(const v8f& dlo, const v8f& dhi, bool hihalf) {
  v16bf r;
#pragma unroll
  for (int v = 0; v < 4; ++v) {           // elements j = 0..7
    float s0 = swapx16(dhi[2*v]);
    float s1 = swapx16(dhi[2*v+1]);
    r[2*v]   = (__bf16)(hihalf ? s0 : dlo[2*v]);
    r[2*v+1] = (__bf16)(hihalf ? s1 : dlo[2*v+1]);
  }
#pragma unroll
  for (int v = 4; v < 8; ++v) {           // elements j = 8..15
    float s0 = swapx16(dlo[2*v-8]);
    float s1 = swapx16(dlo[2*v-7]);
    r[2*v]   = (__bf16)(hihalf ? dhi[2*v-8] : s0);
    r[2*v+1] = (__bf16)(hihalf ? dhi[2*v-7] : s1);
  }
  return r;
}

__device__ __forceinline__ v16bf lds_afrag(const char* base, int f, int lane) {
  return *(const v16bf*)(base + (((size_t)f * 32 + lane) << 5));
}
__device__ __forceinline__ v16bf g_afrag(const unsigned short* w, int f, int lane) {
  return *(const v16bf*)(w + (((size_t)f * 32 + lane) << 4));
}
__device__ __forceinline__ v8f g_bias(const float* b, int tile, int lane) {
  return *(const v8f*)(b + (((size_t)tile * 32 + lane) << 3));
}
__device__ __forceinline__ v8f ld_tile(char* base, int tile, int lane) {
  return *(const v8f*)(base + (((size_t)tile * 32 + lane) << 5));
}
__device__ __forceinline__ void st_tile(char* base, int tile, int lane, v8f v) {
  *(v8f*)(base + (((size_t)tile * 32 + lane) << 5)) = v;
}
__device__ __forceinline__ float sigmoidf_(float x) { return 1.f / (1.f + __expf(-x)); }
__device__ __forceinline__ float tanhf_(float x)    { return 2.f / (1.f + __expf(-2.f * x)) - 1.f; }

// ---------------- pre-pass: pack weights into A-fragment order (bf16) --------
// A-frag element map: lane%16 = row-in-tile, k_local = j + 16*hi
__global__ void pack_afrag_k(const float* __restrict__ src, unsigned short* __restrict__ dst,
                             int MT, int KT, int ld, int transpose) {
  int gid = blockIdx.x * blockDim.x + threadIdx.x;
  int total = MT * KT * 32;
  if (gid >= total) return;
  int lane = gid & 31, f = gid >> 5;
  int kt = f % KT, mt = f / KT;
  int hi = (lane >> 4) & 1;
  int m  = mt * 16 + (lane & 15);
#pragma unroll
  for (int j = 0; j < 16; ++j) {
    int kl = j + 16 * hi;
    int k  = kt * 32 + kl;
    float v = transpose ? src[(size_t)k * ld + m] : src[(size_t)m * ld + k];
    __bf16 bv = (__bf16)v;
    unsigned short u;
    __builtin_memcpy(&u, &bv, 2);
    dst[((size_t)f * 32 + lane) * 16 + j] = u;
  }
}

// bias packed into f32 D-tile order: value = src[tile*16 + e + 8*hi]
__global__ void pack_bias_k(const float* __restrict__ src, float* __restrict__ dst, int tiles) {
  int gid = blockIdx.x * blockDim.x + threadIdx.x;
  if (gid >= tiles * 32) return;
  int lane = gid & 31, tile = gid >> 5;
  int hi = (lane >> 4) & 1;
#pragma unroll
  for (int e = 0; e < 8; ++e)
    dst[((size_t)tile * 32 + lane) * 8 + e] = src[tile * 16 + e + 8 * hi];
}

// ------- main persistent scan: 2 cooperating waves per 16-row batch tile -----
__global__ void __launch_bounds__(64)
odernn_scan(const float* __restrict__ obs, const float* __restrict__ tp,
            const char* __restrict__ wsb, float* __restrict__ out) {
  extern __shared__ char smem[];
  const int  tid    = threadIdx.x;
  const int  lane   = tid & 31;
  const int  wave   = tid >> 5;          // 0 or 1
  const bool hihalf = lane >= 16;
  const int  bIdx   = blockIdx.x * 16 + (lane & 15);

  {
    const uint4* s = (const uint4*)(wsb + OFF_W1T);
    uint4* d = (uint4*)smem;
    for (int i = tid; i < 131072 / 16; i += 64) d[i] = s[i];
  }
  __syncthreads();

  const char* w1lds = smem;
  const char* w2lds = smem + LDS_W2;
  char*       accb  = smem + LDS_ACC + wave * 8192;     // this wave's RK4 accumulator
  char*       xchW  = smem + LDS_XCH + wave * 8192;     // this wave's partial kbuf
  char*       xchP  = smem + LDS_XCH + (wave ^ 1) * 8192;  // partner's partial
  const unsigned short* wih    = (const unsigned short*)(wsb + OFF_WIH);
  const unsigned short* whh    = (const unsigned short*)(wsb + OFF_WHH);
  const unsigned short* wembT  = (const unsigned short*)(wsb + OFF_WEMBT);
  const unsigned short* wprojT = (const unsigned short*)(wsb + OFF_WPROJT);
  const float* b1f    = (const float*)(wsb + OFF_B1F);
  const float* b2f    = (const float*)(wsb + OFF_B2F);
  const float* bihf   = (const float*)(wsb + OFF_BIHF);
  const float* bhhf   = (const float*)(wsb + OFF_BHHF);
  const float* bembf  = (const float*)(wsb + OFF_BEMBF);
  const float* bprojf = (const float*)(wsb + OFF_BPROJF);

  v8f h[8];
#pragma unroll
  for (int i = 0; i < 8; ++i)
#pragma unroll
    for (int e = 0; e < 8; ++e) h[i][e] = 0.f;

  float prev = tp[0];
  for (int t = 0; t < T_STEPS; ++t) {
    float tpt = tp[t];
    float dt = (t == 0) ? 0.f : (tpt - prev);   // block-uniform
    prev = tpt;

    if (dt > 0.f) {                                 // ---- RK4 evolve ----
      const float hs = dt / (float)NODE;
      for (int stp = 0; stp < NODE; ++stp) {
        v8f kbuf[8];
#pragma unroll
        for (int i = 0; i < 8; ++i)
#pragma unroll
          for (int e = 0; e < 8; ++e) kbuf[i][e] = 0.f;

#pragma unroll 1
        for (int s = 0; s < 4; ++s) {
          float c = (s == 0) ? 0.f : ((s == 3) ? hs : 0.5f * hs);
          // y = h + c*k_prev  ->  bf16 B-fragments (half-swap via ds_swizzle)
          v16bf hB[4];
#pragma unroll
          for (int kt = 0; kt < 4; ++kt) {
            v8f ylo, yhi;
#pragma unroll
            for (int e = 0; e < 8; ++e) {
              ylo[e] = h[2*kt][e]   + c * kbuf[2*kt][e];
              yhi[e] = h[2*kt+1][e] + c * kbuf[2*kt+1][e];
            }
            hB[kt] = make_bfrag(ylo, yhi, hihalf);
          }
          // partial k: this wave covers j = 4*wave .. 4*wave+3 of NH
          if (wave == 0) {
#pragma unroll
            for (int m = 0; m < 8; ++m) kbuf[m] = g_bias(b2f, m, lane);
          } else {
#pragma unroll
            for (int m = 0; m < 8; ++m)
#pragma unroll
              for (int e = 0; e < 8; ++e) kbuf[m][e] = 0.f;
          }
#pragma unroll 1
          for (int jj = 0; jj < 4; ++jj) {
            int j = wave * 4 + jj;
            v8f u0 = g_bias(b1f, 2*j,   lane);
            v8f u1 = g_bias(b1f, 2*j+1, lane);
#pragma unroll
            for (int kt = 0; kt < 4; ++kt) {
              u0 = wmma_bf16(lds_afrag(w1lds, (2*j)  *4 + kt, lane), hB[kt], u0);
              u1 = wmma_bf16(lds_afrag(w1lds, (2*j+1)*4 + kt, lane), hB[kt], u1);
            }
#pragma unroll
            for (int e = 0; e < 8; ++e) { u0[e] = fmaxf(u0[e], 0.f); u1[e] = fmaxf(u1[e], 0.f); }
            v16bf uB = make_bfrag(u0, u1, hihalf);
#pragma unroll
            for (int m = 0; m < 8; ++m)
              kbuf[m] = wmma_bf16(lds_afrag(w2lds, m*8 + j, lane), uB, kbuf[m]);
          }
          // sum the two partial k's (both waves end with identical kbuf)
#pragma unroll
          for (int i = 0; i < 8; ++i) st_tile(xchW, i, lane, kbuf[i]);
          __syncthreads();
#pragma unroll
          for (int i = 0; i < 8; ++i) {
            v8f p = ld_tile(xchP, i, lane);
#pragma unroll
            for (int e = 0; e < 8; ++e) kbuf[i][e] += p[e];
          }
          __syncthreads();   // partner may overwrite its region next stage

          // RK4 accumulator in per-wave LDS region
          if (s == 0) {
#pragma unroll
            for (int i = 0; i < 8; ++i) st_tile(accb, i, lane, kbuf[i]);
          } else if (s < 3) {
#pragma unroll
            for (int i = 0; i < 8; ++i) {
              v8f a = ld_tile(accb, i, lane);
#pragma unroll
              for (int e = 0; e < 8; ++e) a[e] += 2.f * kbuf[i][e];
              st_tile(accb, i, lane, a);
            }
          } else {                                 // fold last stage
            const float g = hs / 6.f;
#pragma unroll
            for (int i = 0; i < 8; ++i) {
              v8f a = ld_tile(accb, i, lane);
#pragma unroll
              for (int e = 0; e < 8; ++e) h[i][e] += g * (a[e] + kbuf[i][e]);
            }
          }
        }
      }
    }

    // ---- GRU update (run redundantly by both waves; h stays replicated) ----
    v16bf xB[2];
    {
      const float* orow = obs + ((size_t)bIdx * T_STEPS + t) * OBS_D;
#pragma unroll
      for (int kt = 0; kt < 2; ++kt) {
        v16bf r;
#pragma unroll
        for (int v = 0; v < 8; ++v) {
          int d = kt * 32 + 2 * v + (hihalf ? 16 : 0);
          float2 p = *(const float2*)(orow + d);
          r[2*v] = (__bf16)p.x;  r[2*v+1] = (__bf16)p.y;
        }
        xB[kt] = r;
      }
    }
    v16bf xeB[4], hB2[4];
    {
      v8f xe0, xe1;
#pragma unroll
      for (int kt = 0; kt < 4; ++kt) {
        xe0 = g_bias(bembf, 2*kt, lane);
        xe1 = g_bias(bembf, 2*kt + 1, lane);
#pragma unroll
        for (int q = 0; q < 2; ++q) {
          xe0 = wmma_bf16(g_afrag(wembT, (2*kt)  *2 + q, lane), xB[q], xe0);
          xe1 = wmma_bf16(g_afrag(wembT, (2*kt+1)*2 + q, lane), xB[q], xe1);
        }
        xeB[kt] = make_bfrag(xe0, xe1, hihalf);
        hB2[kt] = make_bfrag(h[2*kt], h[2*kt+1], hihalf);
      }
    }
#pragma unroll
    for (int m = 0; m < 8; ++m) {
      v8f gi, gh, rT, zT;
      // r gate (tiles 0..7)
      gi = g_bias(bihf, m, lane);
      gh = g_bias(bhhf, m, lane);
#pragma unroll
      for (int kt = 0; kt < 4; ++kt) {
        gi = wmma_bf16(g_afrag(wih, m*4 + kt, lane), xeB[kt], gi);
        gh = wmma_bf16(g_afrag(whh, m*4 + kt, lane), hB2[kt], gh);
      }
#pragma unroll
      for (int e = 0; e < 8; ++e) rT[e] = sigmoidf_(gi[e] + gh[e]);
      // z gate (tiles 8..15)
      gi = g_bias(bihf, 8 + m, lane);
      gh = g_bias(bhhf, 8 + m, lane);
#pragma unroll
      for (int kt = 0; kt < 4; ++kt) {
        gi = wmma_bf16(g_afrag(wih, (8+m)*4 + kt, lane), xeB[kt], gi);
        gh = wmma_bf16(g_afrag(whh, (8+m)*4 + kt, lane), hB2[kt], gh);
      }
#pragma unroll
      for (int e = 0; e < 8; ++e) zT[e] = sigmoidf_(gi[e] + gh[e]);
      // n gate (tiles 16..23) + h update
      gi = g_bias(bihf, 16 + m, lane);
      gh = g_bias(bhhf, 16 + m, lane);
#pragma unroll
      for (int kt = 0; kt < 4; ++kt) {
        gi = wmma_bf16(g_afrag(wih, (16+m)*4 + kt, lane), xeB[kt], gi);
        gh = wmma_bf16(g_afrag(whh, (16+m)*4 + kt, lane), hB2[kt], gh);
      }
#pragma unroll
      for (int e = 0; e < 8; ++e) {
        float n = tanhf_(gi[e] + rT[e] * gh[e]);
        float z = zT[e];
        h[m][e] = (1.f - z) * n + z * h[m][e];
      }
    }
  }

  // ---- final projection: outT = w_projT @ hT + b_proj (stores from wave0) ----
  v16bf hBf[4];
#pragma unroll
  for (int kt = 0; kt < 4; ++kt) hBf[kt] = make_bfrag(h[2*kt], h[2*kt+1], hihalf);
#pragma unroll
  for (int m = 0; m < 4; ++m) {
    v8f o = g_bias(bprojf, m, lane);
#pragma unroll
    for (int kt = 0; kt < 4; ++kt)
      o = wmma_bf16(g_afrag(wprojT, m*4 + kt, lane), hBf[kt], o);
    if (wave == 0) {
#pragma unroll
      for (int e = 0; e < 8; ++e) {
        int row = m * 16 + e + (hihalf ? 8 : 0);
        if (row < 32) out[(size_t)bIdx * 32 + row] = o[e];                      // z0_mean
        else out[(size_t)B_TOTAL * 32 + (size_t)bIdx * 32 + (row - 32)] = o[e]; // z0_logvar
      }
    }
  }
}

extern "C" void kernel_launch(void* const* d_in, const int* in_sizes, int n_in,
                              void* d_out, int out_size, void* d_ws, size_t ws_size,
                              hipStream_t stream) {
  (void)in_sizes; (void)n_in; (void)out_size; (void)ws_size;
  const float* obs    = (const float*)d_in[0];
  const float* tp     = (const float*)d_in[1];
  const float* w_emb  = (const float*)d_in[2];
  const float* b_emb  = (const float*)d_in[3];
  const float* w_ih   = (const float*)d_in[4];
  const float* w_hh   = (const float*)d_in[5];
  const float* b_ih   = (const float*)d_in[6];
  const float* b_hh   = (const float*)d_in[7];
  const float* w1     = (const float*)d_in[8];
  const float* b1     = (const float*)d_in[9];
  const float* w2     = (const float*)d_in[10];
  const float* b2     = (const float*)d_in[11];
  const float* w_proj = (const float*)d_in[12];
  const float* b_proj = (const float*)d_in[13];
  char* ws = (char*)d_ws;

  auto pa = [&](const float* s, unsigned int off, int MT, int KT, int ld, int tr) {
    int total = MT * KT * 32;
    pack_afrag_k<<<(total + 255) / 256, 256, 0, stream>>>(
        s, (unsigned short*)(ws + off), MT, KT, ld, tr);
  };
  auto pb = [&](const float* s, unsigned int off, int tiles) {
    int total = tiles * 32;
    pack_bias_k<<<(total + 127) / 128, 128, 0, stream>>>(s, (float*)(ws + off), tiles);
  };
  pa(w1,     OFF_W1T,    16, 4, 256, 1);   // w1^T  (256x128)
  pa(w2,     OFF_W2T,     8, 8, 128, 1);   // w2^T  (128x256)
  pa(w_ih,   OFF_WIH,    24, 4, 128, 0);   // w_ih  (384x128)
  pa(w_hh,   OFF_WHH,    24, 4, 128, 0);   // w_hh  (384x128)
  pa(w_emb,  OFF_WEMBT,   8, 2, 128, 1);   // w_emb^T (128x64)
  pa(w_proj, OFF_WPROJT,  4, 4,  64, 1);   // w_proj^T (64x128)
  pb(b1,     OFF_B1F,   16);
  pb(b2,     OFF_B2F,    8);
  pb(b_ih,   OFF_BIHF,  24);
  pb(b_hh,   OFF_BHHF,  24);
  pb(b_emb,  OFF_BEMBF,  8);
  pb(b_proj, OFF_BPROJF, 4);

  // 256 blocks x 2 cooperating waves (16 batch rows each), 160KB dynamic LDS
  odernn_scan<<<256, 64, LDS_SIZE, stream>>>(obs, tp, (const char*)ws, (float*)d_out);
}
